// SequentialGptOssMLP_3341484556957
// MI455X (gfx1250) — matrix operand order repaired and checked by reading, compiled, not run on hardware
//
#include <hip/hip_runtime.h>
#include <hip/hip_bf16.h>

// Problem constants (from reference): B=2, S=1024 -> T=2048 tokens.
#define T_TOK 2048
#define HDIM  2048
#define IDIM  2048
#define NEXP  16
#define TOPK  4
#define ALPHA_C 1.702f
#define LIMIT_C 7.0f

typedef __attribute__((ext_vector_type(16))) __bf16 v16bf;
typedef __attribute__((ext_vector_type(8)))  __bf16 v8bf;
typedef __attribute__((ext_vector_type(8)))  float  v8f;
typedef __attribute__((ext_vector_type(4)))  int    v4i;

typedef __attribute__((address_space(1))) v4i  as1_v4i;
typedef __attribute__((address_space(3))) v4i  as3_v4i;
typedef __attribute__((address_space(3))) char as3_char;

// fp32 -> bf16: native conversion (compiler emits v_cvt_pk_bf16_f32 pairs)
__device__ __forceinline__ __bf16 f2bf(float f) { return (__bf16)f; }

template <int BASE>
__device__ __forceinline__ void pack_f4(v16bf& d, float4 q0, float4 q1) {
  d[BASE + 0] = f2bf(q0.x); d[BASE + 1] = f2bf(q0.y);
  d[BASE + 2] = f2bf(q0.z); d[BASE + 3] = f2bf(q0.w);
  d[BASE + 4] = f2bf(q1.x); d[BASE + 5] = f2bf(q1.y);
  d[BASE + 6] = f2bf(q1.z); d[BASE + 7] = f2bf(q1.w);
}

template <int BASE>
__device__ __forceinline__ void pack_bf8(v16bf& d, v8bf s) {
  d[BASE + 0] = s[0]; d[BASE + 1] = s[1]; d[BASE + 2] = s[2]; d[BASE + 3] = s[3];
  d[BASE + 4] = s[4]; d[BASE + 5] = s[5]; d[BASE + 6] = s[6]; d[BASE + 7] = s[7];
}

// B-fragment (32x16 bf16): lane n holds K = kbase + hi*16 .. +15, contiguous in memory.
__device__ __forceinline__ v16bf load_w16(const float* __restrict__ wrow) {
  const float4* p = reinterpret_cast<const float4*>(wrow);
  float4 q0 = p[0], q1 = p[1], q2 = p[2], q3 = p[3];
  v16bf b;
  pack_f4<0>(b, q0, q1);
  pack_f4<8>(b, q2, q3);
  return b;
}

// A-fragment (16x32 bf16): lane (m, hi): elems 0..15 = K kbase+hi*16 .. +15, contiguous.
__device__ __forceinline__ v16bf load_a16(const float* __restrict__ xrow, int k, int hi) {
  const float4* p0 = reinterpret_cast<const float4*>(xrow + k + hi * 16);
  const float4* p1 = reinterpret_cast<const float4*>(xrow + k + hi * 16 + 8);
  v16bf a;
  pack_f4<0>(a, p0[0], p0[1]);
  pack_f4<8>(a, p1[0], p1[1]);
  return a;
}

// ---- CDNA5 async copy to LDS (ASYNCcnt-tracked) + split waits ---------------------
__device__ __forceinline__ void async_b128(const void* g, void* l) {
#if __has_builtin(__builtin_amdgcn_global_load_async_to_lds_b128)
  __builtin_amdgcn_global_load_async_to_lds_b128(
      (as1_v4i*)(unsigned long long)(__UINTPTR_TYPE__)g, (as3_v4i*)l, 0, 0);
#else
  unsigned loff = (unsigned)(__UINTPTR_TYPE__)(as3_char*)l;
  unsigned long long ga = (unsigned long long)(__UINTPTR_TYPE__)g;
  asm volatile("global_load_async_to_lds_b128 %0, %1, off" :: "v"(loff), "v"(ga) : "memory");
#endif
}

__device__ __forceinline__ void wait_async0() {
#if __has_builtin(__builtin_amdgcn_s_wait_asynccnt)
  __builtin_amdgcn_s_wait_asynccnt(0);
#else
  asm volatile("s_wait_asynccnt 0x0" ::: "memory");
#endif
}

__device__ __forceinline__ void wait_ds0() {
#if __has_builtin(__builtin_amdgcn_s_wait_dscnt)
  __builtin_amdgcn_s_wait_dscnt(0);
#else
  asm volatile("s_wait_dscnt 0x0" ::: "memory");
#endif
}

// ---------------- Router: logits -> top4 -> softmax -> dense score + expert bins ----
__global__ __launch_bounds__(256) void router_kernel(
    const float* __restrict__ x, const float* __restrict__ rw,
    const float* __restrict__ rb, float* __restrict__ score,
    int* __restrict__ counts, int* __restrict__ tokids,
    float* __restrict__ probs) {
  int wave = threadIdx.x >> 5;
  int lane = threadIdx.x & 31;
  int t = blockIdx.x * 8 + wave;
  const float* xrow = x + (size_t)t * HDIM;

  float acc[NEXP];
#pragma unroll
  for (int e = 0; e < NEXP; ++e) acc[e] = 0.f;
  for (int h = lane; h < HDIM; h += 32) {
    float xv = xrow[h];
#pragma unroll
    for (int e = 0; e < NEXP; ++e) acc[e] = fmaf(xv, rw[e * HDIM + h], acc[e]);
  }
#pragma unroll
  for (int e = 0; e < NEXP; ++e) {
#pragma unroll
    for (int off = 16; off >= 1; off >>= 1) acc[e] += __shfl_xor(acc[e], off, 32);
    acc[e] += rb[e];
  }

  // top-4 (strict > picks lowest index on ties, matching jax top_k)
  float tv[TOPK]; int ti[TOPK]; unsigned used = 0;
#pragma unroll
  for (int kk = 0; kk < TOPK; ++kk) {
    float best = -3.4e38f; int bi = 0;
#pragma unroll
    for (int e = 0; e < NEXP; ++e) {
      bool ok = !((used >> e) & 1u);
      if (ok && acc[e] > best) { best = acc[e]; bi = e; }
    }
    tv[kk] = best; ti[kk] = bi; used |= 1u << bi;
  }
  float p[TOPK]; float s = 0.f;
#pragma unroll
  for (int kk = 0; kk < TOPK; ++kk) { p[kk] = __expf(tv[kk] - tv[0]); s += p[kk]; }
  float inv = 1.f / s;
#pragma unroll
  for (int kk = 0; kk < TOPK; ++kk) p[kk] *= inv;

  if (lane < NEXP) {
    float sc = 0.f;
#pragma unroll
    for (int kk = 0; kk < TOPK; ++kk) if (ti[kk] == lane) sc = p[kk];
    score[(size_t)t * NEXP + lane] = sc;
  }
  if (lane < TOPK) {
    int e = ti[lane];
    float pr = p[lane];
    int pos = atomicAdd(&counts[e], 1);
    tokids[e * T_TOK + pos] = t;
    probs[e * T_TOK + pos] = pr;
  }
}

// ---------------- Exclusive prefix sum over 16 expert counts -----------------------
__global__ void prefix_kernel(const int* __restrict__ counts, int* __restrict__ seg) {
  if (threadIdx.x == 0) {
    int s = 0;
    for (int e = 0; e < NEXP; ++e) { seg[e] = s; s += counts[e]; }
    seg[NEXP] = s;
  }
}

// ---------------- Grouped gate/up GEMM + activation -> bf16 h ----------------------
// Block: 512 thr = 16 waves (4x4), tile M=64 (gathered tokens) x N=64, K loop over H.
__global__ __launch_bounds__(512) void gateup_kernel(
    const float* __restrict__ x, const float* __restrict__ gate_w,
    const float* __restrict__ gate_b, const float* __restrict__ up_w,
    const float* __restrict__ up_b, const int* __restrict__ counts,
    const int* __restrict__ tokids, const int* __restrict__ seg,
    __bf16* __restrict__ h_ws) {
  int e = blockIdx.z;
  int cnt = counts[e];
  int mbase = blockIdx.y * 64;
  if (mbase >= cnt) return;                 // unused tile -> whole-block exit
  int wave = threadIdx.x >> 5, lane = threadIdx.x & 31;
  int wm = wave >> 2, wn = wave & 3;
  int m0 = mbase + wm * 16;
  if (m0 >= cnt) return;                    // wave uniform, no barriers in this kernel
  int nbase = blockIdx.x * 64 + wn * 16;
  int hi = lane >> 4, l15 = lane & 15;

  int arow = m0 + l15;
  int tok = tokids[e * T_TOK + (arow < cnt ? arow : cnt - 1)];
  const float* xrow = x + (size_t)tok * HDIM;
  int n = nbase + l15;
  const float* gw = gate_w + ((size_t)e * IDIM + n) * HDIM;
  const float* uw = up_w + ((size_t)e * IDIM + n) * HDIM;

  v8f cg = {0.f, 0.f, 0.f, 0.f, 0.f, 0.f, 0.f, 0.f};
  v8f cu = {0.f, 0.f, 0.f, 0.f, 0.f, 0.f, 0.f, 0.f};
  for (int k = 0; k < HDIM; k += 32) {
    v16bf a  = load_a16(xrow, k, hi);
    v16bf bg = load_w16(gw + k + hi * 16);
    v16bf bu = load_w16(uw + k + hi * 16);
    cg = __builtin_amdgcn_wmma_f32_16x16x32_bf16(false, a, false, bg, (short)0, cg, false, false);
    cu = __builtin_amdgcn_wmma_f32_16x16x32_bf16(false, a, false, bu, (short)0, cu, false, false);
  }

  float gb = gate_b[(size_t)e * IDIM + n];
  float ub = up_b[(size_t)e * IDIM + n];
  int segb = seg[e];
#pragma unroll
  for (int j = 0; j < 8; ++j) {
    int m = m0 + j + 8 * hi;                // C layout: VGPR j -> M = j + 8*hi
    float g = cg[j] + gb;
    float u = cu[j] + ub;
    u = fminf(fmaxf(u, -LIMIT_C), LIMIT_C);
    g = fminf(g, LIMIT_C);
    // g * sigmoid(alpha*g): raw v_rcp_f32 (~1 ulp), plenty for bf16 storage
    float glu = g * __builtin_amdgcn_rcpf(1.f + __expf(-ALPHA_C * g));
    float hv = (u + 1.f) * glu;
    if (m < cnt) h_ws[(size_t)(segb + m) * IDIM + n] = f2bf(hv);
  }
}

// ---------------- Grouped down GEMM + weighted scatter-add -------------------------
// Block: 512 thr = 16 waves, tile M=64 x N=64, K loop over I.
// A (bf16 h rows) is gathered into LDS with async-to-LDS (double buffered).
__global__ __launch_bounds__(512) void down_kernel(
    const __bf16* __restrict__ h_ws, const float* __restrict__ down_w,
    const float* __restrict__ down_b, const int* __restrict__ counts,
    const int* __restrict__ tokids, const float* __restrict__ probs,
    const int* __restrict__ seg, float* __restrict__ outp) {
  __shared__ __align__(16) char smem[2][64 * 64];   // 2 x (64 rows x 64B K-chunk)

  int e = blockIdx.z;
  int cnt = counts[e];
  int mbase = blockIdx.y * 64;
  if (mbase >= cnt) return;                 // whole-block uniform exit (no barrier yet)
  int wave = threadIdx.x >> 5, lane = threadIdx.x & 31;
  int wm = wave >> 2, wn = wave & 3;
  int m0 = mbase + wm * 16;                 // may exceed cnt: keep running (barriers)
  int nbase = blockIdx.x * 64 + wn * 16;
  int hi = lane >> 4, l15 = lane & 15;
  int segb = seg[e];

  int n = nbase + l15;
  const float* dw = down_w + ((size_t)e * HDIM + n) * IDIM;

  // Async stage of 64 rows x 32 bf16 (64B) of h into smem[buf]; waves 0..7 (256 thr),
  // one B128 per lane, per-lane gathered global addresses.
  int sidx = threadIdx.x;                   // 0..255 participate
  int srow = sidx >> 2, sc = sidx & 3;
  int srowg = mbase + srow; if (srowg >= cnt) srowg = cnt - 1;
  const __bf16* srcrow = h_ws + (size_t)(segb + srowg) * IDIM + sc * 8;

#define STAGE(buf, kk)                                                        \
  do {                                                                        \
    if (wave < 8)                                                             \
      async_b128(srcrow + (kk), &smem[buf][srow * 64 + sc * 16]);             \
  } while (0)

  STAGE(0, 0);

  v8f c = {0.f, 0.f, 0.f, 0.f, 0.f, 0.f, 0.f, 0.f};
  for (int k = 0; k < IDIM; k += 32) {
    int cur = (k >> 5) & 1;
    wait_async0();                          // own async stores to LDS complete
    wait_ds0();                             // own LDS reads of other buffer complete
    __syncthreads();                        // everyone's staging + reads complete
    if (k + 32 < IDIM) STAGE(1 - cur, k + 32);

    const char* abase = &smem[cur][(wm * 16 + l15) * 64 + hi * 32];
    v16bf a;
    pack_bf8<0>(a, *reinterpret_cast<const v8bf*>(abase));
    pack_bf8<8>(a, *reinterpret_cast<const v8bf*>(abase + 16));
    v16bf b = load_w16(dw + k + hi * 16);
    c = __builtin_amdgcn_wmma_f32_16x16x32_bf16(false, a, false, b, (short)0, c, false, false);
  }
#undef STAGE

  float db = down_b[(size_t)e * HDIM + n];
#pragma unroll
  for (int j = 0; j < 8; ++j) {
    int m = m0 + j + 8 * hi;
    if (m < cnt) {
      float pr = probs[e * T_TOK + m];
      int tok = tokids[e * T_TOK + m];
      float val = (c[j] + db) * pr;
      __hip_atomic_fetch_add(outp + (size_t)tok * HDIM + n, val,
                             __ATOMIC_RELAXED, __HIP_MEMORY_SCOPE_AGENT);
    }
  }
}

extern "C" void kernel_launch(void* const* d_in, const int* in_sizes, int n_in,
                              void* d_out, int out_size, void* d_ws, size_t ws_size,
                              hipStream_t stream) {
  (void)in_sizes; (void)n_in; (void)out_size; (void)ws_size;
  const float* x        = (const float*)d_in[0];
  const float* router_w = (const float*)d_in[1];
  const float* router_b = (const float*)d_in[2];
  const float* gate_w   = (const float*)d_in[3];
  const float* gate_b   = (const float*)d_in[4];
  const float* up_w     = (const float*)d_in[5];
  const float* up_b     = (const float*)d_in[6];
  const float* down_w   = (const float*)d_in[7];
  const float* down_b   = (const float*)d_in[8];

  float* out   = (float*)d_out;                       // [T, H] next_states
  float* score = out + (size_t)T_TOK * HDIM;          // [T, E] routing_score

  // Workspace layout:
  //   [0,64)        counts[16]
  //   [128,196)     seg[17]
  //   [256, +128K)  tokids[16][2048]
  //   [+128K,+256K) probs[16][2048]
  //   [+256K, ...)  h bf16 [8192][2048] = 32 MB
  char* ws = (char*)d_ws;
  int*    counts = (int*)ws;
  int*    seg    = (int*)(ws + 128);
  int*    tokids = (int*)(ws + 256);
  float*  probs  = (float*)(ws + 256 + (size_t)NEXP * T_TOK * 4);
  __bf16* h_ws   = (__bf16*)(ws + 256 + 2 * (size_t)NEXP * T_TOK * 4);

  (void)hipMemsetAsync(out, 0, (size_t)T_TOK * HDIM * sizeof(float), stream);
  (void)hipMemsetAsync(ws, 0, 256, stream);

  router_kernel<<<T_TOK / 8, 256, 0, stream>>>(x, router_w, router_b, score,
                                               counts, tokids, probs);
  prefix_kernel<<<1, 32, 0, stream>>>(counts, seg);
  gateup_kernel<<<dim3(IDIM / 64, T_TOK / 64, NEXP), 512, 0, stream>>>(
      x, gate_w, gate_b, up_w, up_b, counts, tokids, seg, h_ws);
  down_kernel<<<dim3(HDIM / 64, T_TOK / 64, NEXP), 512, 0, stream>>>(
      h_ws, down_w, down_b, counts, tokids, probs, seg, out);
}
